// Stage1AIMQuantizer_83227876262440
// MI455X (gfx1250) — matrix operand order repaired and checked by reading, not compiled, not run on hardware
//
#include <hip/hip_runtime.h>
#include <hip/hip_bf16.h>
#include <stdint.h>

// ---------------------------------------------------------------------------
// Stage1 AIM quantizer, fused, for MI455X (gfx1250, wave32, WMMA).
//   out = [ z_q (B*P f32) | indices (B, stored as f32) | loss (1) ]
//
// h = l2norm(z) @ W + b is computed as rin * (bf16(z) @ bf16(W)) + b, with the
// per-row 1/||z|| folded into the epilogue (scale-invariance of fp rounding).
// ---------------------------------------------------------------------------

typedef __attribute__((ext_vector_type(16))) __bf16 v16bf;
typedef __attribute__((ext_vector_type(8)))  float  v8f;

#define B_ROWS  65536
#define D_INF   1408
#define PDIM    256
#define KCB     64
#define BM      64            // rows per block
#define BK      32            // K per WMMA step
#define NKT     (D_INF / BK)  // 44
#define HSTR    260           // padded f32 LDS stride for h

// workspace layout (bytes)
#define WS_WT    0                           // P*D_IN bf16 (transposed W)
#define WS_CB    (WS_WT + PDIM * D_INF * 2)  // K*P bf16 codebook
#define WS_CN2   (WS_CB + KCB * PDIM * 2)    // K f32  |c|^2
#define WS_PART  (WS_CN2 + KCB * 4)          // 1024 f32 loss partials

// LDS layout (bytes) inside dynamic smem
#define L_AS     0                            // 2 * BM*BK u16  (8192, ping-pong)
#define L_HS     (L_AS + 2 * BM * BK * 2)     // BM*HSTR f32    (66560)
#define L_ZS     (L_HS + BM * HSTR * 4)       // BM*PDIM u16    (32768)
#define L_SS     (L_ZS + BM * PDIM * 2)       // BM*KCB f32     (16384)
#define L_INV    (L_SS + BM * KCB * 4)        // BM f32
#define L_IDX    (L_INV + BM * 4)             // BM i32
#define L_RED    (L_IDX + BM * 4)             // 8 f32
#define SMEM_BYTES (L_RED + 32)               // = 124448

union Frag { uint4 u[2]; v16bf v; };

__device__ __forceinline__ uint16_t f2bf(float f) {
  uint32_t u = __float_as_uint(f);
  u += 0x7FFFu + ((u >> 16) & 1u);   // round-to-nearest-even
  return (uint16_t)(u >> 16);
}
__device__ __forceinline__ float bf2f(uint16_t h) {
  return __uint_as_float(((uint32_t)h) << 16);
}

// ---------------------------------------------------------------------------
// prep: W [D_IN][P] f32  ->  Wt [P][D_IN] bf16   (WMMA B-frag friendly)
// ---------------------------------------------------------------------------
__global__ void aimq_prep_wt(const float* __restrict__ W,
                             uint16_t* __restrict__ Wt) {
  int idx = blockIdx.x * 256 + threadIdx.x;   // exactly P*D_IN threads
  int n = idx / D_INF;
  int k = idx - n * D_INF;
  Wt[idx] = f2bf(W[(size_t)k * PDIM + n]);
}

// prep: codebook -> bf16, and |c|^2 per code
__global__ void aimq_prep_cb(const float* __restrict__ cbf,
                             uint16_t* __restrict__ cbu,
                             float* __restrict__ cn2) {
  __shared__ float red[8];
  int b = blockIdx.x, t = threadIdx.x;
  float c = cbf[b * PDIM + t];
  cbu[b * PDIM + t] = f2bf(c);
  float s = c * c;
  for (int off = 16; off; off >>= 1) s += __shfl_xor(s, off);
  if ((t & 31) == 0) red[t >> 5] = s;
  __syncthreads();
  if (t == 0) {
    float tot = 0.f;
    for (int i = 0; i < 8; ++i) tot += red[i];
    cn2[b] = tot;
  }
}

// ---------------------------------------------------------------------------
// main fused kernel: 1024 blocks x 256 threads (8 waves)
// ---------------------------------------------------------------------------
__global__ void __launch_bounds__(256, 2)
aimq_main(const float* __restrict__ z, const float* __restrict__ bias,
          const float* __restrict__ gamma, const float* __restrict__ beta,
          const float* __restrict__ cbf,
          const uint16_t* __restrict__ Wt, const uint16_t* __restrict__ cbu,
          const float* __restrict__ cn2,
          float* __restrict__ out_zq, float* __restrict__ out_idx,
          float* __restrict__ partials) {
  extern __shared__ char smem[];
  uint16_t* As   = (uint16_t*)(smem + L_AS);   // two 64x32 bf16 buffers
  float*    hs   = (float*)   (smem + L_HS);
  uint16_t* zs   = (uint16_t*)(smem + L_ZS);
  float*    Ss   = (float*)   (smem + L_SS);
  float*    invn = (float*)   (smem + L_INV);
  int*      idxs = (int*)     (smem + L_IDX);
  float*    red  = (float*)   (smem + L_RED);

  const int t    = threadIdx.x;
  const int lane = t & 31;
  const int w    = t >> 5;       // wave 0..7
  const int l    = lane & 15;    // lane within half
  const int hi   = lane >> 4;    // half select
  const int mw   = w >> 2;       // M wave group (0..1) -> rows mw*32..+31
  const int nw   = w & 3;        // N wave group (0..3) -> cols nw*64..+63
  const long rowBase = (long)blockIdx.x * BM;

  const int r = t >> 2, q = t & 3;   // row / quad-chunk mapping (4 thr/row)
  const float* asrc = z + (rowBase + r) * D_INF + q * 8;

  // ---------- stage first raw-bf16 A tile into buffer 0 ----------
  {
    float4 v0 = ((const float4*)asrc)[0];
    float4 v1 = ((const float4*)asrc)[1];
    uint16_t tmp[8];
    tmp[0] = f2bf(v0.x); tmp[1] = f2bf(v0.y);
    tmp[2] = f2bf(v0.z); tmp[3] = f2bf(v0.w);
    tmp[4] = f2bf(v1.x); tmp[5] = f2bf(v1.y);
    tmp[6] = f2bf(v1.z); tmp[7] = f2bf(v1.w);
    *((uint4*)&As[r * BK + q * 8]) = *(const uint4*)tmp;
  }

  // ---------- row L2 norms (1/max(||x||, 1e-12)) ----------
  {
    const float4* zr = (const float4*)(z + (rowBase + r) * D_INF) + q * 88;
    float s = 0.f;
#pragma unroll 4
    for (int i = 0; i < 88; ++i) {
      float4 v = zr[i];
      s += v.x * v.x + v.y * v.y + v.z * v.z + v.w * v.w;
    }
    s += __shfl_xor(s, 1, 4);
    s += __shfl_xor(s, 2, 4);
    if (q == 0) invn[r] = 1.f / fmaxf(sqrtf(s), 1e-12f);
  }

  // ---------- GEMM: acc = bf16(z) @ bf16(W), f32 accum (WMMA) ----------
  v8f acc[2][4];
  const v8f vzero = {0.f, 0.f, 0.f, 0.f, 0.f, 0.f, 0.f, 0.f};
#pragma unroll
  for (int a = 0; a < 2; ++a)
#pragma unroll
    for (int bq = 0; bq < 4; ++bq) acc[a][bq] = vzero;

  for (int kt = 0; kt < NKT; ++kt) {
    __syncthreads();   // buffer (kt&1) ready for all; prior reads drained

    // stage next tile into the other buffer (ping-pong, 1 barrier / step)
    if (kt + 1 < NKT) {
      const float* src = asrc + (kt + 1) * BK;
      float4 v0 = ((const float4*)src)[0];
      float4 v1 = ((const float4*)src)[1];
      uint16_t tmp[8];
      tmp[0] = f2bf(v0.x); tmp[1] = f2bf(v0.y);
      tmp[2] = f2bf(v0.z); tmp[3] = f2bf(v0.w);
      tmp[4] = f2bf(v1.x); tmp[5] = f2bf(v1.y);
      tmp[6] = f2bf(v1.z); tmp[7] = f2bf(v1.w);
      *((uint4*)&As[((kt + 1) & 1) * (BM * BK) + r * BK + q * 8]) =
          *(const uint4*)tmp;
      if (kt + 2 < NKT)
        __builtin_prefetch(src + BK, 0, 3);   // near-scope global_prefetch_b8
    }

    // B fragments: Wt is [N][K] bf16 -> 2x16B contiguous global loads/lane
    Frag bf[4];
#pragma unroll
    for (int nf = 0; nf < 4; ++nf) {
      const uint4* p = (const uint4*)(Wt + (size_t)(nw * 64 + nf * 16 + l) * D_INF
                                      + kt * BK + hi * 16);
      bf[nf].u[0] = p[0];
      bf[nf].u[1] = p[1];
    }
    // A fragments from LDS (contiguous K layout: e0-15 K=hi*16.., matches B)
    Frag af[2];
    const uint16_t* abuf = As + (kt & 1) * (BM * BK);
#pragma unroll
    for (int mf = 0; mf < 2; ++mf) {
      const uint16_t* ap = abuf + (mw * 32 + mf * 16 + l) * BK;
      af[mf].u[0] = *(const uint4*)(ap + hi * 16);
      af[mf].u[1] = *(const uint4*)(ap + hi * 16 + 8);
    }
#pragma unroll
    for (int mf = 0; mf < 2; ++mf)
#pragma unroll
      for (int nf = 0; nf < 4; ++nf)
        acc[mf][nf] = __builtin_amdgcn_wmma_f32_16x16x32_bf16(
            false, af[mf].v, false, bf[nf].v, (short)0, acc[mf][nf], false, false);
  }
  __syncthreads();   // invn complete; accumulators done

  // ---------- h = rin*acc + bias -> LDS (C/D layout: elem rr -> M=hi*8+rr, N=l)
#pragma unroll
  for (int mf = 0; mf < 2; ++mf)
#pragma unroll
    for (int nf = 0; nf < 4; ++nf) {
      const int N = nw * 64 + nf * 16 + l;
      const float bv = bias[N];
#pragma unroll
      for (int rr = 0; rr < 8; ++rr) {
        const int M = mw * 32 + mf * 16 + hi * 8 + rr;
        hs[M * HSTR + N] = acc[mf][nf][rr] * invn[M] + bv;
      }
    }
  __syncthreads();

  // ---------- LayerNorm + L2 normalize -> zs (bf16) ----------
  {
    const float* hrow = hs + r * HSTR + q * 64;
    float sum = 0.f, sq = 0.f;
#pragma unroll 8
    for (int c = 0; c < 64; ++c) { float x = hrow[c]; sum += x; sq += x * x; }
    sum += __shfl_xor(sum, 1, 4); sum += __shfl_xor(sum, 2, 4);
    sq  += __shfl_xor(sq, 1, 4);  sq  += __shfl_xor(sq, 2, 4);
    const float mu   = sum * (1.f / 256.f);
    const float var  = sq * (1.f / 256.f) - mu * mu;
    const float rstd = rsqrtf(var + 1e-5f);
    const float* gp = gamma + q * 64;
    const float* bp = beta + q * 64;
    float s2 = 0.f;
#pragma unroll 8
    for (int c = 0; c < 64; ++c) {
      float zp = (hrow[c] - mu) * rstd * gp[c] + bp[c];
      s2 += zp * zp;
    }
    s2 += __shfl_xor(s2, 1, 4); s2 += __shfl_xor(s2, 2, 4);
    const float rl2 = 1.f / fmaxf(sqrtf(s2), 1e-12f);
    uint16_t* zrow = zs + r * PDIM + q * 64;
#pragma unroll 8
    for (int c = 0; c < 64; ++c) {
      float zp = (hrow[c] - mu) * rstd * gp[c] + bp[c];
      zrow[c] = f2bf(zp * rl2);
    }
  }
  __syncthreads();

  // ---------- distance GEMM: dot = z @ codebook^T via WMMA ----------
  {
    const int n0 = nw * 16;     // 16 codes per N-wave-group
    v8f dacc[2];
    dacc[0] = vzero; dacc[1] = vzero;
#pragma unroll
    for (int kk = 0; kk < 8; ++kk) {
      Frag cb;
      const uint4* cp = (const uint4*)(cbu + (size_t)(n0 + l) * PDIM
                                       + kk * 32 + hi * 16);
      cb.u[0] = cp[0];
      cb.u[1] = cp[1];
#pragma unroll
      for (int mf = 0; mf < 2; ++mf) {
        Frag zf;
        const uint16_t* zp = zs + (mw * 32 + mf * 16 + l) * PDIM + kk * 32;
        zf.u[0] = *(const uint4*)(zp + hi * 16);
        zf.u[1] = *(const uint4*)(zp + hi * 16 + 8);
        dacc[mf] = __builtin_amdgcn_wmma_f32_16x16x32_bf16(
            false, zf.v, false, cb.v, (short)0, dacc[mf], false, false);
      }
    }
    const float cn = cn2[n0 + l];   // |c|^2 ; |z|^2 == 1 after L2 norm
#pragma unroll
    for (int mf = 0; mf < 2; ++mf)
#pragma unroll
      for (int rr = 0; rr < 8; ++rr) {
        const int M = mw * 32 + mf * 16 + hi * 8 + rr;
        Ss[M * KCB + (n0 + l)] = 1.f - 2.f * dacc[mf][rr] + cn;
      }
  }
  __syncthreads();

  // ---------- argmin over 64 codes ----------
  if (t < BM) {
    const float* sr = Ss + t * KCB;
    float best = sr[0];
    int bi = 0;
    for (int c = 1; c < KCB; ++c) {
      float v = sr[c];
      if (v < best) { best = v; bi = c; }
    }
    idxs[t] = bi;
    out_idx[rowBase + t] = (float)bi;   // indices stored as f32 in d_out
  }
  __syncthreads();

  // ---------- gather z_q (exact f32 codebook) + commit-loss partial ----------
  {
    const int code = idxs[r];
    const float* crow = cbf + (size_t)code * PDIM + q * 64;
    float* orow = out_zq + (rowBase + r) * PDIM + q * 64;
    const uint16_t* zrow = zs + r * PDIM + q * 64;
    float ls = 0.f;
#pragma unroll 4
    for (int c = 0; c < 64; ++c) {
      float cv = crow[c];
      orow[c] = cv;                     // z_q_st == z_q numerically
      float d = cv - bf2f(zrow[c]);
      ls += d * d;
    }
    for (int off = 16; off; off >>= 1) ls += __shfl_xor(ls, off);
    if (lane == 0) red[w] = ls;
  }
  __syncthreads();
  if (t == 0) {
    float s = 0.f;
    for (int i = 0; i < 8; ++i) s += red[i];
    partials[blockIdx.x] = s;           // deterministic per-block slot
  }
}

// fixed-order reduction of loss partials
__global__ void aimq_finalize(const float* __restrict__ partials,
                              float* __restrict__ out_loss) {
  __shared__ float red[8];
  int t = threadIdx.x;
  float s = partials[t] + partials[t + 256] + partials[t + 512] + partials[t + 768];
  for (int off = 16; off; off >>= 1) s += __shfl_xor(s, off);
  if ((t & 31) == 0) red[t >> 5] = s;
  __syncthreads();
  if (t == 0) {
    float tot = 0.f;
    for (int i = 0; i < 8; ++i) tot += red[i];
    *out_loss = 0.25f * tot / ((float)B_ROWS * (float)PDIM);
  }
}

extern "C" void kernel_launch(void* const* d_in, const int* in_sizes, int n_in,
                              void* d_out, int out_size, void* d_ws, size_t ws_size,
                              hipStream_t stream) {
  const float* z     = (const float*)d_in[0];
  const float* W     = (const float*)d_in[1];
  const float* bias  = (const float*)d_in[2];
  const float* gamma = (const float*)d_in[3];
  const float* beta  = (const float*)d_in[4];
  const float* cbf   = (const float*)d_in[5];

  char* ws = (char*)d_ws;
  uint16_t* Wt   = (uint16_t*)(ws + WS_WT);
  uint16_t* cbu  = (uint16_t*)(ws + WS_CB);
  float*    cn2  = (float*)(ws + WS_CN2);
  float*    part = (float*)(ws + WS_PART);

  float* out      = (float*)d_out;
  float* out_zq   = out;
  float* out_idx  = out + (size_t)B_ROWS * PDIM;
  float* out_loss = out + (size_t)B_ROWS * PDIM + B_ROWS;

  aimq_prep_wt<<<dim3(D_INF), dim3(256), 0, stream>>>(W, Wt);
  aimq_prep_cb<<<dim3(KCB), dim3(256), 0, stream>>>(cbf, cbu, cn2);

  (void)hipFuncSetAttribute((const void*)aimq_main,
                            hipFuncAttributeMaxDynamicSharedMemorySize,
                            (int)SMEM_BYTES);
  aimq_main<<<dim3(B_ROWS / BM), dim3(256), SMEM_BYTES, stream>>>(
      z, bias, gamma, beta, cbf, Wt, cbu, cn2, out_zq, out_idx, part);

  aimq_finalize<<<dim3(1), dim3(256), 0, stream>>>(part, out_loss);
}
